// ResRnn_49100066128076
// MI455X (gfx1250) — compile-verified, matrix operand order checked by reading, not run on hardware
//
#include <hip/hip_runtime.h>
#include <hip/hip_bf16.h>

#define SEQ 512
#define BATCH 64
#define IN_W 256
#define STREAM_W 1024
#define OUT_W 256
#define LIN 0.99999f

#define NWG 64
#define BLOCK 128
#define SPAD 1032   // bf16 elements per LDS row (1024 + 8 pad -> rotates LDS banks)

typedef __attribute__((ext_vector_type(16))) __bf16 bf16x16;
typedef __attribute__((ext_vector_type(8)))  float  f32x8;

union Frag { uint4 q[2]; bf16x16 v; };

// native f32 -> bf16 (RNE) — lets the backend use v_cvt instead of bit-twiddling
__device__ __forceinline__ __bf16 f2bf(float f) { return static_cast<__bf16>(f); }

__device__ __forceinline__ unsigned pack2bf(float a, float b) {
    unsigned short lo = __builtin_bit_cast(unsigned short, f2bf(a));
    unsigned short hi = __builtin_bit_cast(unsigned short, f2bf(b));
    return (unsigned)lo | ((unsigned)hi << 16);
}

__device__ __forceinline__ void grid_barrier(unsigned* cnt, unsigned* gen) {
    __syncthreads();
    if (threadIdx.x == 0) {
        unsigned g = __hip_atomic_load(gen, __ATOMIC_ACQUIRE, __HIP_MEMORY_SCOPE_AGENT);
        unsigned prev = __hip_atomic_fetch_add(cnt, 1u, __ATOMIC_ACQ_REL, __HIP_MEMORY_SCOPE_AGENT);
        if (prev + 1u == (unsigned)NWG) {
            __hip_atomic_store(cnt, 0u, __ATOMIC_RELAXED, __HIP_MEMORY_SCOPE_AGENT);
            __hip_atomic_fetch_add(gen, 1u, __ATOMIC_ACQ_REL, __HIP_MEMORY_SCOPE_AGENT);
        } else {
            while (__hip_atomic_load(gen, __ATOMIC_ACQUIRE, __HIP_MEMORY_SCOPE_AGENT) == g) {
                __builtin_amdgcn_s_sleep(2);
            }
        }
    }
    __syncthreads();
}

__global__ void resrnn_bar_init(char* ws) {
    unsigned* bar = (unsigned*)(ws + (5u << 20));
    if (threadIdx.x == 0) { bar[0] = 0u; bar[1] = 0u; }
}

__global__ __launch_bounds__(BLOCK, 1)
void resrnn_persistent(const float* __restrict__ x,
                       const float* __restrict__ init_stream,
                       const int*   __restrict__ perm,
                       const float* __restrict__ W1, const float* __restrict__ b1,
                       const float* __restrict__ W2, const float* __restrict__ b2,
                       float* __restrict__ out, char* __restrict__ ws) {
    // workspace carve
    __bf16*   w1bf = (__bf16*)(ws);                              // 2 MiB
    __bf16*   w2bf = (__bf16*)(ws + (2u << 20));                 // 2 MiB
    float*    buf0 = (float*) (ws + (4u << 20));                 // 256 KiB
    float*    buf1 = (float*) (ws + (4u << 20) + (256u << 10));  // 256 KiB
    __bf16*   hbf  = (__bf16*)(ws + (4u << 20) + (512u << 10));  // 128 KiB
    unsigned* bar  = (unsigned*)(ws + (5u << 20));

    const int tid      = threadIdx.x;
    const int gtid     = blockIdx.x * BLOCK + tid;
    const int nthreads = NWG * BLOCK;
    const int mi   = blockIdx.x & 3;    // M strip (16 rows of the 64-row batch)
    const int cb   = blockIdx.x >> 2;   // 64-column block (0..15)
    const int wave = tid >> 5;          // 4 wave32 per WG
    const int lane = tid & 31;
    const int ln15 = lane & 15;
    const int half = lane >> 4;

    __shared__ __bf16 s_lds[16 * SPAD];
    __shared__ int    perm_lds[STREAM_W];

    // ---- one-time init: weights -> bf16, stream0 = (1-lin)*initial_stream ----
    for (int i = gtid; i < STREAM_W * STREAM_W; i += nthreads) {
        w1bf[i] = f2bf(W1[i]);
        w2bf[i] = f2bf(W2[i]);
    }
    for (int i = gtid; i < BATCH * STREAM_W; i += nthreads) {
        buf0[i] = (1.0f - LIN) * init_stream[i & (STREAM_W - 1)];
    }
    // cache permutation in LDS (loop-invariant across all 512 steps)
    for (int i = tid; i < STREAM_W; i += BLOCK) perm_lds[i] = perm[i];
    grid_barrier(&bar[0], &bar[1]);

    const int tile_n = cb * 4 + wave;        // 0..63
    const int col    = tile_n * 16 + ln15;   // output column this lane owns
    const float bias1 = b1[col];
    const float bias2 = b2[col];
    const int   pcol  = perm_lds[col];

    for (int t = 0; t < SEQ; ++t) {
        const float* cur = (t & 1) ? buf1 : buf0;
        float*       nxt = (t & 1) ? buf0 : buf1;
        const float* xt  = x + (size_t)t * BATCH * IN_W;

        // ---- s strip (16 x 1024) -> LDS bf16 (permuted gather), 2 cols/iter ----
        for (int p = tid; p < 16 * (STREAM_W / 2); p += BLOCK) {
            int r  = p >> 9;                  // 0..15
            int jp = (p & 511) * 2;           // even column
            int row = mi * 16 + r;
            int pj0 = perm_lds[jp];
            int pj1 = perm_lds[jp + 1];
            float v0 = cur[row * STREAM_W + pj0];
            float v1 = cur[row * STREAM_W + pj1];
            if (pj0 < IN_W) v0 += xt[row * IN_W + pj0];
            if (pj1 < IN_W) v1 += xt[row * IN_W + pj1];
            *reinterpret_cast<unsigned*>(&s_lds[r * SPAD + jp]) = pack2bf(v0, v1);
        }
        __syncthreads();

        // ---- GEMM1: h[mi, tile_n] = relu(s @ W1^T + b1) ----
        f32x8 acc = {};
        const __bf16* arow = &s_lds[ln15 * SPAD];
        const __bf16* brow = w1bf + (size_t)col * STREAM_W;
        #pragma unroll 4
        for (int kk = 0; kk < STREAM_W; kk += 32) {
            Frag fa, fb;
            fa.q[0] = *reinterpret_cast<const uint4*>(arow + kk + half * 8);
            fa.q[1] = *reinterpret_cast<const uint4*>(arow + kk + 16 + half * 8);
            fb.q[0] = *reinterpret_cast<const uint4*>(brow + kk + half * 8);
            fb.q[1] = *reinterpret_cast<const uint4*>(brow + kk + 16 + half * 8);
            acc = __builtin_amdgcn_wmma_f32_16x16x32_bf16(
                      false, fa.v, false, fb.v, (short)0, acc, false, false);
        }
        {
            const int rbase = mi * 16 + half * 8;
            #pragma unroll
            for (int v = 0; v < 8; ++v) {
                float hv = acc[v] + bias1;
                hv = hv > 0.0f ? hv : 0.0f;
                hbf[(rbase + v) * STREAM_W + col] = f2bf(hv);
            }
        }
        grid_barrier(&bar[0], &bar[1]);

        // ---- GEMM2: y = h @ W2^T + b2 ; new = lin*s + (1-lin)*y ----
        f32x8 acc2 = {};
        const __bf16* arow2 = hbf + (size_t)(mi * 16 + ln15) * STREAM_W;
        const __bf16* brow2 = w2bf + (size_t)col * STREAM_W;
        #pragma unroll 4
        for (int kk = 0; kk < STREAM_W; kk += 32) {
            Frag fa, fb;
            fa.q[0] = *reinterpret_cast<const uint4*>(arow2 + kk + half * 8);
            fa.q[1] = *reinterpret_cast<const uint4*>(arow2 + kk + 16 + half * 8);
            fb.q[0] = *reinterpret_cast<const uint4*>(brow2 + kk + half * 8);
            fb.q[1] = *reinterpret_cast<const uint4*>(brow2 + kk + 16 + half * 8);
            acc2 = __builtin_amdgcn_wmma_f32_16x16x32_bf16(
                       false, fa.v, false, fb.v, (short)0, acc2, false, false);
        }
        {
            const int rbase = mi * 16 + half * 8;
            #pragma unroll
            for (int v = 0; v < 8; ++v) {
                int row = rbase + v;
                float sv = cur[row * STREAM_W + pcol];       // exact f32 s path
                if (pcol < IN_W) sv += xt[row * IN_W + pcol];
                float y = acc2[v] + bias2;
                nxt[row * STREAM_W + col] = LIN * sv + (1.0f - LIN) * y;
            }
        }
        grid_barrier(&bar[0], &bar[1]);
    }

    // ---- outputs: [ last[:, 768:1024] (64x256) , last (64x1024) ] ----
    const float* last = buf0;  // after 512 steps, ping-pong lands in buf0
    const int total = BATCH * OUT_W + BATCH * STREAM_W;
    for (int i = gtid; i < total; i += nthreads) {
        if (i < BATCH * OUT_W) {
            int b = i >> 8, c = i & (OUT_W - 1);
            out[i] = last[b * STREAM_W + (STREAM_W - OUT_W) + c];
        } else {
            out[i] = last[i - BATCH * OUT_W];
        }
    }
}

extern "C" void kernel_launch(void* const* d_in, const int* in_sizes, int n_in,
                              void* d_out, int out_size, void* d_ws, size_t ws_size,
                              hipStream_t stream) {
    const float* x           = (const float*)d_in[0];
    const float* init_stream = (const float*)d_in[1];
    const int*   perm        = (const int*)  d_in[2];
    const float* W1          = (const float*)d_in[3];
    const float* b1          = (const float*)d_in[4];
    const float* W2          = (const float*)d_in[5];
    const float* b2          = (const float*)d_in[6];
    float*       out         = (float*)d_out;
    char*        ws          = (char*)d_ws;

    resrnn_bar_init<<<1, 32, 0, stream>>>(ws);
    resrnn_persistent<<<NWG, BLOCK, 0, stream>>>(x, init_stream, perm,
                                                 W1, b1, W2, b2, out, ws);
}
